// CoPredictor_35948876268304
// MI455X (gfx1250) — compile-verified, matrix operand order checked by reading, not run on hardware
//
#include <hip/hip_runtime.h>
#include <hip/hip_bf16.h>
#include <math.h>

// ---------------------------------------------------------------------------
// CDNA5 (gfx1250) wave32 WMMA bf16 implementation of CoPredictor.
//   B=2, N=512, HID=768, BIAF=256, CLS=14, SIZE_DIM=25, N_POS=30
// All matmuls via v_wmma_f32_16x16x32_bf16 (bf16 in, f32 accum), with
// 2x2 / 2x1 register tiling so each WMMA amortizes its fragment loads.
// ---------------------------------------------------------------------------

typedef __attribute__((ext_vector_type(16))) __bf16 v16bf;
typedef __attribute__((ext_vector_type(8)))  __bf16 v8bf;
typedef __attribute__((ext_vector_type(8)))  float  v8f;
typedef __attribute__((ext_vector_type(4)))  float  f4v;

#define KPAD 288   // 257 (BIAF+1) padded up to multiple of 32

// WMMA wrapper: D = A*B + C  (bf16 x bf16 -> f32)
__device__ inline v8f wmma_bf16(v16bf a, v16bf b, v8f c) {
  return __builtin_amdgcn_wmma_f32_16x16x32_bf16(
      /*neg_a=*/false, a, /*neg_b=*/false, b,
      /*c_mod=*/(short)0, c, /*reuse_a=*/false, /*reuse_b=*/false);
}

// Load 16 bf16 A-operand elements for one lane from an f32 row (aligned case).
// Lane (group g = lane>>4) holds K = [k0, k0+8) and [k0+16, k0+24), k0 = kk + g*8.
__device__ inline v16bf ld16_f32(const float* __restrict__ row, int k0) {
  v16bf a;
  const f4v* p0 = (const f4v*)(row + k0);
  const f4v* p1 = (const f4v*)(row + k0 + 16);
  f4v x0 = p0[0], x1 = p0[1], y0 = p1[0], y1 = p1[1];
#pragma unroll
  for (int i = 0; i < 4; i++) {
    a[i]      = (__bf16)x0[i];
    a[4 + i]  = (__bf16)x1[i];
    a[8 + i]  = (__bf16)y0[i];
    a[12 + i] = (__bf16)y1[i];
  }
  return a;
}

// Same but with per-element index clamp (unaligned / short rows like biaf_W).
__device__ inline v16bf ld16_f32_clamp(const float* __restrict__ row, int k0, int kmax) {
  v16bf a;
#pragma unroll
  for (int i = 0; i < 8; i++) {
    int j0 = k0 + i;       if (j0 > kmax) j0 = kmax;
    int j1 = k0 + 16 + i;  if (j1 > kmax) j1 = kmax;
    a[i]     = (__bf16)row[j0];
    a[8 + i] = (__bf16)row[j1];
  }
  return a;
}

// Load 16 bf16 A-operand elements from a bf16 row (16B-aligned runs of 8).
__device__ inline v16bf ld16_bf(const __bf16* __restrict__ row, int k0) {
  v16bf a;
  v8bf lo = *(const v8bf*)(row + k0);
  v8bf hi = *(const v8bf*)(row + k0 + 16);
#pragma unroll
  for (int i = 0; i < 8; i++) { a[i] = lo[i]; a[8 + i] = hi[i]; }
  return a;
}

// ---------------------------------------------------------------------------
// Kernel 0: padding init: h1/t1 col 256 = 1.0, cols 257..287 = 0;
//           zero u's unwritten tail columns 272..287.
// ---------------------------------------------------------------------------
__global__ void init_pad_kernel(__bf16* __restrict__ h1, __bf16* __restrict__ t1,
                                __bf16* __restrict__ u) {
  int i = blockIdx.x * blockDim.x + threadIdx.x;
  if (i < 1024 * 32) {                         // 2*512 rows x 32 pad cols
    int row = i >> 5, c = i & 31;
    __bf16 v = (c == 0) ? (__bf16)1.0f : (__bf16)0.0f;
    h1[(size_t)row * KPAD + 256 + c] = v;
    t1[(size_t)row * KPAD + 256 + c] = v;
  }
  if (i < 28672 * 16) {                        // 2*14*1024 rows x 16 pad cols (u has 2*14*512)
    int row = i >> 4, c = i & 15;
    u[(size_t)row * KPAD + 272 + c] = (__bf16)0.0f;
  }
}

// ---------------------------------------------------------------------------
// Kernel 1: four projection GEMMs (1024 x 256 x 768).
// One wave computes a 32x32 macro-tile (2x2 WMMA tiles): 4 WMMAs per K-step
// reuse 4 fragment loads (halves the f32->bf16 convert work per WMMA).
//   sel 0: h = gelu(x @ mlp1_w^T + b)  -> h1 bf16 (stride KPAD)
//   sel 1: t = gelu(y @ mlp2_w^T + b)  -> t1 bf16
//   sel 2: head = lrelu(x @ head_w^T)  -> head f32 (stride 256)
//   sel 3: tail = lrelu(x @ tail_w^T)  -> tail f32
// ---------------------------------------------------------------------------
__global__ __launch_bounds__(128) void proj_kernel(
    const float* __restrict__ x, const float* __restrict__ y,
    const float* __restrict__ w0, const float* __restrict__ b0,
    const float* __restrict__ w1, const float* __restrict__ b1,
    const float* __restrict__ w2, const float* __restrict__ b2,
    const float* __restrict__ w3, const float* __restrict__ b3,
    __bf16* __restrict__ h1, __bf16* __restrict__ t1,
    float* __restrict__ headp, float* __restrict__ tailp) {
  int wave = threadIdx.x >> 5, lane = threadIdx.x & 31;
  int id  = blockIdx.x * 4 + wave;       // 0..255 macro tiles
  int sel = blockIdx.y;                  // 0..3
  int tm = id >> 3;                      // 32 macro M-tiles (32 rows each)
  int tn = id & 7;                       // 8 macro N-tiles (32 cols each)
  int g = lane >> 4, l = lane & 15;

  const float* A  = (sel == 1) ? y : x;
  const float* W  = (sel == 0) ? w0 : (sel == 1) ? w1 : (sel == 2) ? w2 : w3;
  const float* Bs = (sel == 0) ? b0 : (sel == 1) ? b1 : (sel == 2) ? b2 : b3;

  const float* arow0 = A + (size_t)(tm * 32 + l) * 768;
  const float* arow1 = arow0 + (size_t)16 * 768;
  const float* brow0 = W + (size_t)(tn * 32 + l) * 768;  // B = A-layout of W rows
  const float* brow1 = brow0 + (size_t)16 * 768;

  v8f c00 = {}, c01 = {}, c10 = {}, c11 = {};
  for (int kk = 0; kk < 768; kk += 32) {
    int k0 = kk + g * 8;
    v16bf a0 = ld16_f32(arow0, k0);
    v16bf a1 = ld16_f32(arow1, k0);
    v16bf bm0 = ld16_f32(brow0, k0);
    v16bf bm1 = ld16_f32(brow1, k0);
    c00 = wmma_bf16(a0, bm0, c00);
    c01 = wmma_bf16(a0, bm1, c01);
    c10 = wmma_bf16(a1, bm0, c10);
    c11 = wmma_bf16(a1, bm1, c11);
  }

#pragma unroll
  for (int mt = 0; mt < 2; mt++) {
#pragma unroll
    for (int nt = 0; nt < 2; nt++) {
      v8f c = (mt == 0) ? (nt == 0 ? c00 : c01) : (nt == 0 ? c10 : c11);
      int n = tn * 32 + nt * 16 + l;     // output feature
      float bias = Bs[n];
#pragma unroll
      for (int r = 0; r < 8; r++) {
        int m = tm * 32 + mt * 16 + r + 8 * g;  // global row (b*N + n_row)
        float v = c[r] + bias;
        if (sel < 2) {
          float gv = 0.5f * v * (1.0f + erff(v * 0.70710678f));  // exact gelu
          (sel == 0 ? h1 : t1)[(size_t)m * KPAD + n] = (__bf16)gv;
        } else {
          float lv = v > 0.0f ? v : 0.01f * v;                   // leaky relu
          (sel == 2 ? headp : tailp)[(size_t)m * 256 + n] = lv;
        }
      }
    }
  }
}

// ---------------------------------------------------------------------------
// Kernel 2: small epilogue tables.
//   rh[b,k,m] = head1[b,m,:] . W[k, 0:257]     (ones bias folded)
//   rt[b,k,n] = tail1[b,n,:] . W[k, 257:514]
//   sdot[k,p] = size_emb[p,:] . W[k, 514:539]  (only 30 distinct rel-positions)
// ---------------------------------------------------------------------------
__global__ void small_kernel(const float* __restrict__ headp,
                             const float* __restrict__ tailp,
                             const float* __restrict__ W,
                             const float* __restrict__ size_emb,
                             float* __restrict__ rh, float* __restrict__ rt,
                             float* __restrict__ sdot) {
  int i = blockIdx.x * blockDim.x + threadIdx.x;
  if (i < 2 * 14 * 512) {
    int m = i & 511, k = (i >> 9) % 14, b = i / (14 * 512);
    const float* hp = headp + (size_t)(b * 512 + m) * 256;
    const float* tp = tailp + (size_t)(b * 512 + m) * 256;
    const float* wh = W + k * 539;
    const float* wt = wh + 257;
    float s0 = wh[256], s1 = wt[256];       // ones column
    for (int h = 0; h < 256; h++) { s0 += hp[h] * wh[h]; s1 += tp[h] * wt[h]; }
    rh[i] = s0; rt[i] = s1;
  }
  if (i < 14 * 30) {
    int k = i / 30, p = i % 30;
    const float* ws = W + k * 539 + 514;
    const float* se = size_emb + p * 25;
    float s = 0.0f;
    for (int h = 0; h < 25; h++) s += se[h] * ws[h];
    sdot[i] = s;
  }
}

// ---------------------------------------------------------------------------
// Kernel 3: u[b,k,n,i] = sum_j t1[b,n,j] * biaf_W[k,i,j]   (512 x 272 x 288)
// One wave computes a 32x16 macro-tile (2 M-tiles sharing one B fragment):
// halves the scalar clamped biaf_W loads per WMMA.
// ---------------------------------------------------------------------------
__global__ __launch_bounds__(128) void biaf_u_kernel(
    const __bf16* __restrict__ t1, const float* __restrict__ biaf_W,
    __bf16* __restrict__ u) {
  int wave = threadIdx.x >> 5, lane = threadIdx.x & 31;
  int id = blockIdx.x * 4 + wave;
  if (id >= 16 * 17) return;
  int bk = blockIdx.y;                    // b*14 + k
  int k = bk % 14, b = bk / 14;
  int tm = id / 17;                       // 16 macro n-row tiles (32 rows each)
  int tn = id % 17;                       // 17 i-tiles (272 cols)
  int g = lane >> 4, l = lane & 15;

  const __bf16* arow0 = t1 + (size_t)(b * 512 + tm * 32 + l) * KPAD;
  const __bf16* arow1 = arow0 + (size_t)16 * KPAD;
  int irow = tn * 16 + l; if (irow > 256) irow = 256;   // stay in-bounds
  const float* brow = biaf_W + (size_t)k * 257 * 257 + (size_t)irow * 257;

  v8f c0 = {}, c1 = {};
  for (int kk = 0; kk < KPAD; kk += 32) {
    int k0 = kk + g * 8;
    v16bf bm = ld16_f32_clamp(brow, k0, 256);   // j clamped; t1 pad cols zero
    v16bf a0 = ld16_bf(arow0, k0);
    v16bf a1 = ld16_bf(arow1, k0);
    c0 = wmma_bf16(a0, bm, c0);
    c1 = wmma_bf16(a1, bm, c1);
  }

  int i = tn * 16 + l;                    // i < 272
  __bf16* ur = u + ((size_t)bk * 512 + tm * 32) * KPAD;
#pragma unroll
  for (int r = 0; r < 8; r++) {
    int nrow = r + 8 * g;
    ur[(size_t)nrow * KPAD + i]              = (__bf16)c0[r];
    ur[(size_t)(nrow + 16) * KPAD + i]       = (__bf16)c1[r];
  }
}

// ---------------------------------------------------------------------------
// Kernel 4: out[b,k,m,n] = h1[b,m,:] . u[b,k,n,:] + rh + rt + sdot[k, rel(n,m)]
// One wave computes a 32x32 macro-tile (2x2 WMMA tiles): 4 WMMAs per K-step
// from 4 fragment loads -> half the L2 traffic of the 1-tile version.
// ---------------------------------------------------------------------------
__global__ __launch_bounds__(128) void biaf_out_kernel(
    const __bf16* __restrict__ h1, const __bf16* __restrict__ u,
    const float* __restrict__ rh, const float* __restrict__ rt,
    const float* __restrict__ sdot, float* __restrict__ out) {
  int wave = threadIdx.x >> 5, lane = threadIdx.x & 31;
  int id = blockIdx.x * 4 + wave;         // 0..255 macro tiles
  int bk = blockIdx.y;
  int k = bk % 14;
  int b = bk / 14;
  int tm = id >> 4, tn = id & 15;         // 16x16 macro tiles of 32x32
  int g = lane >> 4, l = lane & 15;

  const __bf16* arow0 = h1 + (size_t)(b * 512 + tm * 32 + l) * KPAD;
  const __bf16* arow1 = arow0 + (size_t)16 * KPAD;
  const __bf16* brow0 = u + ((size_t)bk * 512 + tn * 32 + l) * KPAD;
  const __bf16* brow1 = brow0 + (size_t)16 * KPAD;

  v8f c00 = {}, c01 = {}, c10 = {}, c11 = {};
  for (int kk = 0; kk < KPAD; kk += 32) {
    int k0 = kk + g * 8;
    v16bf a0 = ld16_bf(arow0, k0);
    v16bf a1 = ld16_bf(arow1, k0);
    v16bf bm0 = ld16_bf(brow0, k0);
    v16bf bm1 = ld16_bf(brow1, k0);
    c00 = wmma_bf16(a0, bm0, c00);
    c01 = wmma_bf16(a0, bm1, c01);
    c10 = wmma_bf16(a1, bm0, c10);
    c11 = wmma_bf16(a1, bm1, c11);
  }

  const float* rhp = rh + (size_t)bk * 512;
  const float* rtp = rt + (size_t)bk * 512;
#pragma unroll
  for (int mt = 0; mt < 2; mt++) {
#pragma unroll
    for (int nt = 0; nt < 2; nt++) {
      v8f c = (mt == 0) ? (nt == 0 ? c00 : c01) : (nt == 0 ? c10 : c11);
      int n = tn * 32 + nt * 16 + l;
      float rtv = rtp[n];
#pragma unroll
      for (int r = 0; r < 8; r++) {
        int m = tm * 32 + mt * 16 + r + 8 * g;
        int p = n - m; if (p < -15) p = -15; if (p > 14) p = 14; p += 15;
        float val = c[r] + rhp[m] + rtv + sdot[k * 30 + p];
        out[((size_t)bk * 512 + m) * 512 + n] = val;
      }
    }
  }
}

// ---------------------------------------------------------------------------
// Launch
// ---------------------------------------------------------------------------
extern "C" void kernel_launch(void* const* d_in, const int* in_sizes, int n_in,
                              void* d_out, int out_size, void* d_ws, size_t ws_size,
                              hipStream_t stream) {
  const float* x       = (const float*)d_in[0];
  const float* y       = (const float*)d_in[1];
  // d_in[2] = z : unused by the reference
  const float* mlp1_w  = (const float*)d_in[3];
  const float* mlp1_b  = (const float*)d_in[4];
  const float* mlp2_w  = (const float*)d_in[5];
  const float* mlp2_b  = (const float*)d_in[6];
  const float* head_w  = (const float*)d_in[7];
  const float* head_b  = (const float*)d_in[8];
  const float* tail_w  = (const float*)d_in[9];
  const float* tail_b  = (const float*)d_in[10];
  const float* biaf_W  = (const float*)d_in[11];
  const float* W       = (const float*)d_in[12];
  const float* size_emb= (const float*)d_in[13];
  float* out = (float*)d_out;

  // workspace layout (all offsets 256B-aligned)
  char* ws = (char*)d_ws;
  __bf16* h1   = (__bf16*)(ws + 0);          //  2*512*288 bf16 = 589824 B
  __bf16* t1   = (__bf16*)(ws + 589824);     //  589824 B
  float*  headp= (float*) (ws + 1179648);    //  2*512*256 f32 = 1048576 B
  float*  tailp= (float*) (ws + 2228224);    //  1048576 B
  __bf16* u    = (__bf16*)(ws + 3276800);    //  2*14*512*288 bf16 = 8257536 B
  float*  rh   = (float*) (ws + 11534336);   //  57344 B
  float*  rt   = (float*) (ws + 11591680);   //  57344 B
  float*  sdot = (float*) (ws + 11649024);   //  1680 B

  init_pad_kernel<<<1792, 256, 0, stream>>>(h1, t1, u);

  proj_kernel<<<dim3(64, 4), 128, 0, stream>>>(
      x, y, mlp1_w, mlp1_b, mlp2_w, mlp2_b, head_w, head_b, tail_w, tail_b,
      h1, t1, headp, tailp);

  small_kernel<<<56, 256, 0, stream>>>(headp, tailp, W, size_emb, rh, rt, sdot);

  biaf_u_kernel<<<dim3(68, 28), 128, 0, stream>>>(t1, biaf_W, u);

  biaf_out_kernel<<<dim3(64, 28), 128, 0, stream>>>(h1, u, rh, rt, sdot, out);
}
